// PointMambaCls_60962765799986
// MI455X (gfx1250) — matrix operand order repaired and checked by reading, not compiled, not run on hardware
//
#include <hip/hip_runtime.h>
#include <hip/hip_bf16.h>

#define B_  8
#define L_  4096
#define D_  384
#define DI_ 768
#define S_  16
#define DTR_ 24
#define KCONV_ 4
#define NC_ 40
#define EPS_ 1e-5f
#define XDBL_LD 64   // x_dbl row stride (56 padded to 64)

typedef __attribute__((ext_vector_type(16))) __bf16        v16bf;
typedef __attribute__((ext_vector_type(8)))  float         v8f;
typedef __attribute__((ext_vector_type(8)))  unsigned int  v8u;

// ---------- helpers ----------
__device__ __forceinline__ unsigned short f2bf(float f) {
    unsigned int u = __builtin_bit_cast(unsigned int, f);
    u += 0x7FFFu + ((u >> 16) & 1u);           // round-to-nearest-even
    return (unsigned short)(u >> 16);
}
__device__ __forceinline__ float bf2f(unsigned short s) {
    return __builtin_bit_cast(float, ((unsigned int)s) << 16);
}

// load a 16-element bf16 fragment as two aligned 16B vectors
__device__ __forceinline__ v16bf load_frag32(const unsigned short* __restrict__ p0,
                                             const unsigned short* __restrict__ p1) {
    const uint4 lo = *(const uint4*)p0;
    const uint4 hi = *(const uint4*)p1;
    v8u u;
    u[0] = lo.x; u[1] = lo.y; u[2] = lo.z; u[3] = lo.w;
    u[4] = hi.x; u[5] = hi.y; u[6] = hi.z; u[7] = hi.w;
    return __builtin_bit_cast(v16bf, u);
}

// ---------- B-matrix swizzle: f32 [K,NSRC] -> bf16 fragment-major ----------
// dst layout: [(kstep, ntile, lane), e] with fragment element e <-> 
// K = kstep*32 + (lane>=16)*8 + (e&7) + (e>>3)*16, N = ntile*16 + (lane&15). Pads N to NSRC<N.
template <int K, int N, int NSRC>
__global__ void k_swizzle_b(const float* __restrict__ src,
                            unsigned short* __restrict__ dst) {
    const int i = blockIdx.x * 256 + threadIdx.x;   // K*N total
    const int e    = i & 15;
    const int ln   = (i >> 4) & 31;
    const int rest = i >> 9;
    const int nt   = rest % (N / 16);
    const int kt   = rest / (N / 16);
    const int k = kt * 32 + (ln >> 4) * 8 + (e & 7) + (e >> 3) * 16;
    const int n = nt * 16 + (ln & 15);
    dst[i] = (n < NSRC) ? f2bf(src[k * NSRC + n]) : (unsigned short)0;
}

// ---------- RMSNorm: x[32768,384] -> xn bf16 ----------
__global__ void k_rmsnorm(const float* __restrict__ x,
                          const float* __restrict__ ln_w,
                          unsigned short* __restrict__ xn) {
    __shared__ float red[128];
    const size_t tok = blockIdx.x;          // 32768 tokens
    const int t = threadIdx.x;              // 128 threads, 3 elems each
    const float* row = x + tok * D_;
    float v0 = row[t], v1 = row[t + 128], v2 = row[t + 256];
    red[t] = v0 * v0 + v1 * v1 + v2 * v2;
    __syncthreads();
    for (int s = 64; s > 0; s >>= 1) {
        if (t < s) red[t] += red[t + s];
        __syncthreads();
    }
    const float inv = rsqrtf(red[0] * (1.0f / D_) + EPS_);
    unsigned short* orow = xn + tok * D_;
    orow[t]       = f2bf(v0 * inv * ln_w[t]);
    orow[t + 128] = f2bf(v1 * inv * ln_w[t + 128]);
    orow[t + 256] = f2bf(v2 * inv * ln_w[t + 256]);
}

// ---------- bf16 WMMA GEMM, compile-time shapes, no guards ----------
// C[M,N] = A[M,K] * B[K,N]; A row-major bf16, B pre-swizzled fragment-major.
// Wave tile: (16*MT) x (16*NT); MT A-frags + NT B-frags -> MT*NT WMMAs per
// k-step, all operands via aligned b128 loads. 8 waves/block along M.
template <int M, int N, int K, int MT, int NT>
__global__ __launch_bounds__(256) void
k_gemm_bf16_t(const unsigned short* __restrict__ A,
              const unsigned short* __restrict__ Bsw,
              float* __restrict__ C) {
    constexpr int NTILES = N / 16;
    const int wave = threadIdx.x >> 5;
    const int lane = threadIdx.x & 31;
    const int m0  = (blockIdx.y * 8 + wave) * (16 * MT);
    const int nt0 = blockIdx.x * NT;
    const int col = lane & 15;
    const int kbA = (lane >> 4) * 8;           // A half-lane K-interleave
    v8f c[MT][NT];
#pragma unroll
    for (int mt = 0; mt < MT; ++mt)
#pragma unroll
        for (int nt = 0; nt < NT; ++nt) c[mt][nt] = (v8f){};
    size_t arow[MT];
#pragma unroll
    for (int mt = 0; mt < MT; ++mt)
        arow[mt] = (size_t)(m0 + mt * 16 + col) * K;

    for (int k0 = 0; k0 < K; k0 += 32) {
        v16bf a[MT];
#pragma unroll
        for (int mt = 0; mt < MT; ++mt)
            a[mt] = load_frag32(A + arow[mt] + k0 + kbA,
                                A + arow[mt] + k0 + 16 + kbA);
#pragma unroll
        for (int nt = 0; nt < NT; ++nt) {
            const unsigned short* bb =
                Bsw + (((size_t)(k0 >> 5) * NTILES + nt0 + nt) * 32 + lane) * 16;
            const v16bf b = load_frag32(bb, bb + 8);
#pragma unroll
            for (int mt = 0; mt < MT; ++mt)
                c[mt][nt] = __builtin_amdgcn_wmma_f32_16x16x32_bf16(
                                false, a[mt], false, b, (short)0, c[mt][nt],
                                false, false);
        }
    }
    const int nr = lane & 15;
    const int mb = (lane >> 4) * 8;
#pragma unroll
    for (int mt = 0; mt < MT; ++mt)
#pragma unroll
        for (int nt = 0; nt < NT; ++nt)
#pragma unroll
            for (int j = 0; j < 8; ++j)
                C[(size_t)(m0 + mt * 16 + mb + j) * N + (nt0 + nt) * 16 + nr] =
                    c[mt][nt][j];
}

// ---------- causal depthwise conv (K=4) + SiLU -> xc bf16 ----------
__global__ void k_conv_silu(const float* __restrict__ xz,
                            const float* __restrict__ conv_w,
                            const float* __restrict__ conv_b,
                            unsigned short* __restrict__ xc) {
    const int d = blockIdx.x * 256 + threadIdx.x;  // 768
    const int l = blockIdx.y;
    const int b = blockIdx.z;
    const size_t tokb = (size_t)b * L_;
    float acc = conv_b[d];
#pragma unroll
    for (int k = 0; k < KCONV_; ++k) {
        const int ls = l - (KCONV_ - 1) + k;
        if (ls >= 0)
            acc += conv_w[d * KCONV_ + k] * xz[(tokb + ls) * (2 * DI_) + d];
    }
    const float s = acc / (1.0f + __expf(-acc));
    xc[(tokb + l) * DI_ + d] = f2bf(s);
}

// ---------- fused selective scan ----------
// grid (3, 8): 256 channels x 8 batches. Per step: x_dbl row staged in LDS;
// dt = softplus(x_dbl[:24] . W_dt[:,d] + bias); 16-state recurrence;
// y gated by SiLU(z) and reduced over L into a register -> ysum[b,d].
__global__ void k_scan(const float* __restrict__ xdbl,      // [B*L,64] (56 used)
                       const unsigned short* __restrict__ xc,// [B*L,768] bf16
                       const float* __restrict__ xz,         // [B*L,1536]
                       const float* __restrict__ W_dt,       // [24,768]
                       const float* __restrict__ dt_bias,    // [768]
                       const float* __restrict__ A_log,      // [768,16]
                       const float* __restrict__ Dskip,      // [768]
                       float* __restrict__ ysum) {           // [B,768]
    const int b = blockIdx.y;
    const int d = blockIdx.x * 256 + threadIdx.x;
    __shared__ float row[DTR_ + 2 * S_];   // 56 floats
    float wdt[DTR_];
#pragma unroll
    for (int r = 0; r < DTR_; ++r) wdt[r] = W_dt[r * DI_ + d];
    float a[S_];
#pragma unroll
    for (int s = 0; s < S_; ++s) a[s] = -__expf(A_log[d * S_ + s]);
    float h[S_];
#pragma unroll
    for (int s = 0; s < S_; ++s) h[s] = 0.0f;
    const float dbias = dt_bias[d];
    const float dsk = Dskip[d];
    float acc = 0.0f;
    const size_t tok0 = (size_t)b * L_;
    for (int l = 0; l < L_; ++l) {
        const size_t tok = tok0 + l;
        __syncthreads();
        if (threadIdx.x < DTR_ + 2 * S_)
            row[threadIdx.x] = xdbl[tok * XDBL_LD + threadIdx.x];
        __syncthreads();
        float dtv = dbias;
#pragma unroll
        for (int r = 0; r < DTR_; ++r) dtv += row[r] * wdt[r];
        dtv = (dtv > 20.0f) ? dtv : log1pf(__expf(dtv));   // softplus
        const float xv = bf2f(xc[tok * DI_ + d]);
        const float dtx = dtv * xv;
        float y = 0.0f;
#pragma unroll
        for (int s = 0; s < S_; ++s) {
            const float dA = __expf(dtv * a[s]);
            h[s] = dA * h[s] + dtx * row[DTR_ + s];
            y += h[s] * row[DTR_ + S_ + s];
        }
        y += xv * dsk;
        const float zv = xz[tok * (2 * DI_) + DI_ + d];
        y *= zv / (1.0f + __expf(-zv));                     // * SiLU(z)
        acc += y;
    }
    ysum[b * DI_ + d] = acc;
}

// ---------- sum over L of residual stream x ----------
__global__ void k_pool_x(const float* __restrict__ x, float* __restrict__ xsum) {
    const int b = blockIdx.y;
    const int dm = blockIdx.x * 128 + threadIdx.x;   // 384
    float acc = 0.0f;
    const size_t base = (size_t)b * L_;
    for (int l = 0; l < L_; ++l) acc += x[(base + l) * D_ + dm];
    xsum[b * D_ + dm] = acc;
}

// ---------- pooled = xsum/L + (ysum/L) @ W_out ----------
__global__ void k_head_pool(const float* __restrict__ xsum,
                            const float* __restrict__ ysum,
                            const float* __restrict__ W_out,
                            float* __restrict__ pooled) {
    const int b = blockIdx.x;
    const int dm = threadIdx.x;                      // 384
    __shared__ float ys[DI_];
    for (int i = threadIdx.x; i < DI_; i += 384) ys[i] = ysum[b * DI_ + i];
    __syncthreads();
    float acc = 0.0f;
    for (int d = 0; d < DI_; ++d) acc += ys[d] * W_out[d * D_ + dm];
    pooled[b * D_ + dm] = (xsum[b * D_ + dm] + acc) * (1.0f / L_);
}

// ---------- FC + BatchNorm(batch stats) + ReLU + classifier ----------
__global__ void k_head_cls(const float* __restrict__ pooled,
                           const float* __restrict__ W_fc,
                           const float* __restrict__ b_fc,
                           const float* __restrict__ gamma,
                           const float* __restrict__ beta,
                           const float* __restrict__ W_cls,
                           const float* __restrict__ b_cls,
                           float* __restrict__ out) {
    __shared__ float hn[B_][256];
    const int t = threadIdx.x;                       // 320 threads
    if (t < 256) {
        float hv[B_];
#pragma unroll
        for (int b = 0; b < B_; ++b) {
            float acc = b_fc[t];
            for (int k = 0; k < D_; ++k)
                acc += pooled[b * D_ + k] * W_fc[k * 256 + t];
            hv[b] = acc;
        }
        float mu = 0.0f, m2 = 0.0f;
#pragma unroll
        for (int b = 0; b < B_; ++b) { mu += hv[b]; m2 += hv[b] * hv[b]; }
        mu *= (1.0f / B_); m2 *= (1.0f / B_);
        const float inv = rsqrtf(m2 - mu * mu + EPS_);
#pragma unroll
        for (int b = 0; b < B_; ++b) {
            float v = (hv[b] - mu) * inv * gamma[t] + beta[t];
            hn[b][t] = v > 0.0f ? v : 0.0f;
        }
    }
    __syncthreads();
    if (t < B_ * NC_) {
        const int b = t / NC_, c = t % NC_;
        float acc = b_cls[c];
        for (int k = 0; k < 256; ++k) acc += hn[b][k] * W_cls[k * NC_ + c];
        out[b * NC_ + c] = acc;
    }
}

extern "C" void kernel_launch(void* const* d_in, const int* in_sizes, int n_in,
                              void* d_out, int out_size, void* d_ws, size_t ws_size,
                              hipStream_t stream) {
    const float* x       = (const float*)d_in[0];
    const float* ln_w    = (const float*)d_in[1];
    const float* W_in    = (const float*)d_in[2];
    const float* conv_w  = (const float*)d_in[3];
    const float* conv_b  = (const float*)d_in[4];
    const float* W_xproj = (const float*)d_in[5];
    const float* W_dt    = (const float*)d_in[6];
    const float* dt_bias = (const float*)d_in[7];
    const float* A_log   = (const float*)d_in[8];
    const float* Dskip   = (const float*)d_in[9];
    const float* W_out   = (const float*)d_in[10];
    const float* W_fc    = (const float*)d_in[11];
    const float* b_fc    = (const float*)d_in[12];
    const float* bn_g    = (const float*)d_in[13];
    const float* bn_b    = (const float*)d_in[14];
    const float* W_cls   = (const float*)d_in[15];
    const float* b_cls   = (const float*)d_in[16];
    float* out = (float*)d_out;

    const int NT_TOK = B_ * L_;             // 32768 tokens
    char* ws = (char*)d_ws;
    size_t off = 0;
    auto carve = [&](size_t bytes) -> void* {
        void* p = (void*)(ws + off);
        off += (bytes + 255) & ~(size_t)255;
        return p;
    };
    unsigned short* xn_bf  = (unsigned short*)carve((size_t)NT_TOK * D_ * 2);
    unsigned short* Win_sw = (unsigned short*)carve((size_t)D_ * 2 * DI_ * 2);
    unsigned short* Wxp_sw = (unsigned short*)carve((size_t)DI_ * XDBL_LD * 2);
    float*          xz     = (float*)carve((size_t)NT_TOK * 2 * DI_ * 4);
    unsigned short* xc_bf  = (unsigned short*)carve((size_t)NT_TOK * DI_ * 2);
    float*          xdbl   = (float*)carve((size_t)NT_TOK * XDBL_LD * 4);
    float*          ysum   = (float*)carve((size_t)B_ * DI_ * 4);
    float*          xsum   = (float*)carve((size_t)B_ * D_ * 4);
    float*          pooled = (float*)carve((size_t)B_ * D_ * 4);

    // 1. weight converts -> fragment-major swizzled bf16 (W_xproj padded to 64)
    k_swizzle_b<D_, 2 * DI_, 2 * DI_>
        <<<(D_ * 2 * DI_) / 256, 256, 0, stream>>>(W_in, Win_sw);
    k_swizzle_b<DI_, XDBL_LD, 56>
        <<<(DI_ * XDBL_LD) / 256, 256, 0, stream>>>(W_xproj, Wxp_sw);
    // 2. RMSNorm
    k_rmsnorm<<<NT_TOK, 128, 0, stream>>>(x, ln_w, xn_bf);
    // 3. GEMM1: [32768,384]x[384,1536] -> xz   (wave tile 32x64, 8 WMMA/step)
    k_gemm_bf16_t<32768, 2 * DI_, D_, 2, 4>
        <<<dim3((2 * DI_) / 64, NT_TOK / 256), 256, 0, stream>>>(xn_bf, Win_sw, xz);
    // 4. conv + SiLU -> xc bf16
    k_conv_silu<<<dim3(DI_ / 256, L_, B_), 256, 0, stream>>>(xz, conv_w, conv_b, xc_bf);
    // 5. GEMM2: [32768,768]x[768,64] -> x_dbl (padded, guard-free)
    k_gemm_bf16_t<32768, XDBL_LD, DI_, 2, 4>
        <<<dim3(1, NT_TOK / 256), 256, 0, stream>>>(xc_bf, Wxp_sw, xdbl);
    // 6. fused selective scan -> ysum
    k_scan<<<dim3(DI_ / 256, B_), 256, 0, stream>>>(
        xdbl, xc_bf, xz, W_dt, dt_bias, A_log, Dskip, ysum);
    // 7. residual pooling + head
    k_pool_x<<<dim3(D_ / 128, B_), 128, 0, stream>>>(x, xsum);
    k_head_pool<<<B_, D_, 0, stream>>>(xsum, ysum, W_out, pooled);
    k_head_cls<<<1, 320, 0, stream>>>(pooled, W_fc, b_fc, bn_g, bn_b, W_cls, b_cls, out);
}
